// MySupernetBertAttention_5514738008566
// MI455X (gfx1250) — matrix operand order repaired and checked by reading, not compiled, not run on hardware
//
#include <hip/hip_runtime.h>
#include <math.h>

typedef _Float16 f16;
typedef __attribute__((ext_vector_type(16))) _Float16 v16h;
typedef __attribute__((ext_vector_type(8)))  _Float16 v8h;
typedef __attribute__((ext_vector_type(2)))  _Float16 h2;
typedef __attribute__((ext_vector_type(8)))  float    v8f;

union V16 { v16h v; v8h h[2]; };

__device__ __forceinline__ v8f wmma16(v16h a, v16h b, v8f c) {
  return __builtin_amdgcn_wmma_f32_16x16x32_f16(false, a, false, b, (short)0, c,
                                                false, false);
}

#define DD   1024
#define SS   512
#define HH   16
#define DHH  64
#define MM   4096   // B*S
#define LDA  72     // LDS stride (f16) for 64-wide tiles: 144B = 36 banks, conflict-free

#define USE_ASYNC_COPY 1

// ---------------------------------------------------------------------------
// Kernel 1: fused QKV projection.  X[4096,1024] @ W[1024,1024] + b
// Block tile 64x64, k-step 64, 8 waves (4x2), each wave owns two 16x16 C
// tiles -> 4 wmma per barrier pair.  Staging uses packed f16 pairs (b32 DS).
// Q,K written f16 [b,h,s,d]; V written transposed f16 [b,h,d,s].
// ---------------------------------------------------------------------------
__global__ __launch_bounds__(256)
void qkv_gemm(const float* __restrict__ X,
              const float* __restrict__ Wq, const float* __restrict__ bq,
              const float* __restrict__ Wk, const float* __restrict__ bk,
              const float* __restrict__ Wv, const float* __restrict__ bv,
              f16* __restrict__ Q, f16* __restrict__ K, f16* __restrict__ Vt)
{
  __shared__ __align__(16) f16 As[64 * LDA];   // row-major   [m][k]
  __shared__ __align__(16) f16 Bs[64 * LDA];   // column-major[n][k]

  const int m0 = blockIdx.x * 64;
  const int n0 = blockIdx.y * 64;
  const int which = blockIdx.z;
  const float* W    = (which == 0) ? Wq : (which == 1) ? Wk : Wv;
  const float* bias = (which == 0) ? bq : (which == 1) ? bk : bv;

  const int tid  = threadIdx.x;
  const int lane = tid & 31, wave = tid >> 5;
  const int col  = lane & 15, half = lane >> 4;
  const int wm   = wave >> 1, wn = wave & 1;

  v8f acc0 = {}, acc1 = {};

  for (int k0 = 0; k0 < DD; k0 += 64) {
    { // stage A 64x64: float2 -> packed h2, one b32 DS store per pair
      const int kp = tid & 31, mb = tid >> 5;
      #pragma unroll
      for (int j = 0; j < 8; ++j) {
        const int m = mb + 8 * j;
        const float2 xv = *(const float2*)(X + (size_t)(m0 + m) * DD + k0 + 2 * kp);
        h2 p; p.x = (f16)xv.x; p.y = (f16)xv.y;
        *(h2*)&As[m * LDA + 2 * kp] = p;
      }
    }
    { // stage B 64x64 column-major: two k-adjacent f32 -> packed h2
      const int n = tid & 63, kb4 = tid >> 6;
      #pragma unroll
      for (int j = 0; j < 8; ++j) {
        const int k = 2 * (kb4 + 4 * j);
        const float lo = W[(size_t)(k0 + k) * DD + n0 + n];
        const float hi = W[(size_t)(k0 + k + 1) * DD + n0 + n];
        h2 p; p.x = (f16)lo; p.y = (f16)hi;
        *(h2*)&Bs[n * LDA + k] = p;
      }
    }
    __syncthreads();

    #pragma unroll
    for (int ks = 0; ks < 2; ++ks) {
      V16 a;
      const f16* ap = &As[(wm * 16 + col) * LDA + ks * 32];
      a.h[0] = *(const v8h*)(ap + half * 8);
      a.h[1] = *(const v8h*)(ap + 16 + half * 8);
      #pragma unroll
      for (int jn = 0; jn < 2; ++jn) {
        const f16* bp = &Bs[(wn * 16 + jn * 32 + col) * LDA + ks * 32];
        V16 b;
        b.h[0] = *(const v8h*)(bp + half * 8);
        b.h[1] = *(const v8h*)(bp + 16 + half * 8);
        if (jn == 0) acc0 = wmma16(a.v, b.v, acc0);
        else         acc1 = wmma16(a.v, b.v, acc1);
      }
    }
    __syncthreads();
  }

  #pragma unroll
  for (int jn = 0; jn < 2; ++jn) {
    const v8f& acc = jn ? acc1 : acc0;
    #pragma unroll
    for (int r = 0; r < 8; ++r) {
      const int gm = m0 + wm * 16 + 8 * half + r;        // (b, s)
      const int gn = n0 + wn * 16 + jn * 32 + col;       // (h, d)
      const int bi = gm >> 9, s = gm & 511;
      const int h  = gn >> 6, d = gn & 63;
      const float v = acc[r] + bias[gn];
      if (which == 0)
        Q[((size_t)(bi * HH + h) * SS + s) * DHH + d] = (f16)v;
      else if (which == 1)
        K[((size_t)(bi * HH + h) * SS + s) * DHH + d] = (f16)v;
      else
        Vt[((size_t)(bi * HH + h) * DHH + d) * SS + s] = (f16)v;
    }
  }
}

// ---------------------------------------------------------------------------
// Kernel 2: flash-style attention per (b,h).  8 waves, 16 q-rows per wave,
// keys streamed in chunks of 32.  P tile bounces through a wave-private LDS
// strip (no barriers needed: same-wave DS ops are in-order on CDNA5).
// ---------------------------------------------------------------------------
__global__ __launch_bounds__(256)
void attn_fa(const f16* __restrict__ Q, const f16* __restrict__ Kf,
             const f16* __restrict__ Vt, const float* __restrict__ mask,
             float* __restrict__ scores, f16* __restrict__ Ctx)
{
  __shared__ __align__(16) f16 Ps[8][16 * 40];

  const int bh = blockIdx.x;
  const int b = bh >> 4, h = bh & 15;
  const int tid  = threadIdx.x;
  const int lane = tid & 31, wave = tid >> 5;
  const int col  = lane & 15, half = lane >> 4;
  const int qbase = blockIdx.y * 128 + wave * 16;

  const f16* qp = Q + ((size_t)bh * SS + qbase + col) * DHH;
  V16 a0, a1;
  a0.h[0] = *(const v8h*)(qp + half * 16);
  a0.h[1] = *(const v8h*)(qp + half * 16 + 8);
  a1.h[0] = *(const v8h*)(qp + 32 + half * 16);
  a1.h[1] = *(const v8h*)(qp + 32 + half * 16 + 8);

  v8f c0 = {}, c1 = {}, c2 = {}, c3 = {};
  float rmax[8], rsum[8];
  #pragma unroll
  for (int r = 0; r < 8; ++r) { rmax[r] = -1e30f; rsum[r] = 0.f; }

  float* srow = scores + ((size_t)bh * SS + qbase) * SS;
  const float LOG2E = 1.44269504088896f;

  for (int kb = 0; kb < SS; kb += 32) {
    v8f s0 = {}, s1 = {};
    {
      const f16* kp = Kf + ((size_t)bh * SS + kb + col) * DHH;
      V16 b0, b1;
      b0.h[0] = *(const v8h*)(kp + half * 16);
      b0.h[1] = *(const v8h*)(kp + half * 16 + 8);
      b1.h[0] = *(const v8h*)(kp + 32 + half * 16);
      b1.h[1] = *(const v8h*)(kp + 32 + half * 16 + 8);
      s0 = wmma16(a0.v, b0.v, s0);
      s0 = wmma16(a1.v, b1.v, s0);
    }
    {
      const f16* kp = Kf + ((size_t)bh * SS + kb + 16 + col) * DHH;
      V16 b0, b1;
      b0.h[0] = *(const v8h*)(kp + half * 16);
      b0.h[1] = *(const v8h*)(kp + half * 16 + 8);
      b1.h[0] = *(const v8h*)(kp + 32 + half * 16);
      b1.h[1] = *(const v8h*)(kp + 32 + half * 16 + 8);
      s1 = wmma16(a0.v, b0.v, s1);
      s1 = wmma16(a1.v, b1.v, s1);
    }

    const float madd0 = mask[b * SS + kb + col] * -10000.0f;
    const float madd1 = mask[b * SS + kb + 16 + col] * -10000.0f;

    float p0[8], p1[8];
    #pragma unroll
    for (int r = 0; r < 8; ++r) {
      const float x0 = s0[r] * 0.125f + madd0;
      const float x1 = s1[r] * 0.125f + madd1;
      srow[(size_t)(8 * half + r) * SS + kb + col]      = x0;  // required output
      srow[(size_t)(8 * half + r) * SS + kb + 16 + col] = x1;
      p0[r] = x0; p1[r] = x1;

      float v = fmaxf(x0, x1);
      v = fmaxf(v, __shfl_xor(v, 1, 16));
      v = fmaxf(v, __shfl_xor(v, 2, 16));
      v = fmaxf(v, __shfl_xor(v, 4, 16));
      v = fmaxf(v, __shfl_xor(v, 8, 16));
      const float nm = fmaxf(rmax[r], v);
      const float f  = exp2f((rmax[r] - nm) * LOG2E);
      rmax[r] = nm;
      rsum[r] *= f;
      c0[r] *= f; c1[r] *= f; c2[r] *= f; c3[r] *= f;

      p0[r] = exp2f((p0[r] - nm) * LOG2E);
      p1[r] = exp2f((p1[r] - nm) * LOG2E);
      float u = p0[r] + p1[r];
      u += __shfl_xor(u, 1, 16);
      u += __shfl_xor(u, 2, 16);
      u += __shfl_xor(u, 4, 16);
      u += __shfl_xor(u, 8, 16);
      rsum[r] += u;

      Ps[wave][(8 * half + r) * 40 + col]      = (f16)p0[r];
      Ps[wave][(8 * half + r) * 40 + 16 + col] = (f16)p1[r];
    }
    // wave-private LDS strip: same-wave DS ops are in-order; just stop the
    // compiler from reordering the loads above the stores.
    asm volatile("" ::: "memory");

    V16 pa;
    const f16* pp = &Ps[wave][col * 40];
    pa.h[0] = *(const v8h*)(pp + half * 16);
    pa.h[1] = *(const v8h*)(pp + half * 16 + 8);

    #pragma unroll
    for (int t4 = 0; t4 < 4; ++t4) {
      const f16* vp = Vt + ((size_t)bh * DHH + t4 * 16 + col) * SS + kb;
      V16 vb;
      vb.h[0] = *(const v8h*)(vp + half * 16);
      vb.h[1] = *(const v8h*)(vp + half * 16 + 8);
      v8f& c = (t4 == 0) ? c0 : (t4 == 1) ? c1 : (t4 == 2) ? c2 : c3;
      c = wmma16(pa.v, vb.v, c);
    }
    asm volatile("" ::: "memory");
  }

  #pragma unroll
  for (int r = 0; r < 8; ++r) {
    const float inv = 1.0f / rsum[r];
    const int qrow = qbase + 8 * half + r;
    const size_t base = ((size_t)b * SS + qrow) * DD + h * DHH;
    Ctx[base + 0 * 16 + col] = (f16)(c0[r] * inv);
    Ctx[base + 1 * 16 + col] = (f16)(c1[r] * inv);
    Ctx[base + 2 * 16 + col] = (f16)(c2[r] * inv);
    Ctx[base + 3 * 16 + col] = (f16)(c3[r] * inv);
  }
}

// ---------------------------------------------------------------------------
// Kernel 3: output projection + bias + residual.  A (Ctx) is already f16, so
// its LDS staging uses CDNA5 async global->LDS (ASYNCcnt) - 16B per lane
// straight into LDS, no VGPR round trip.
// ---------------------------------------------------------------------------
__global__ __launch_bounds__(256)
void out_gemm(const f16* __restrict__ Ctx, const float* __restrict__ Wo,
              const float* __restrict__ bo, const float* __restrict__ X,
              float* __restrict__ Y)
{
  __shared__ __align__(16) f16 As[64 * LDA];
  __shared__ __align__(16) f16 Bs[64 * LDA];

  const int m0 = blockIdx.x * 64;
  const int n0 = blockIdx.y * 64;
  const int tid  = threadIdx.x;
  const int lane = tid & 31, wave = tid >> 5;
  const int col  = lane & 15, half = lane >> 4;
  const int wm   = wave >> 1, wn = wave & 1;

  v8f acc0 = {}, acc1 = {};

  for (int k0 = 0; k0 < DD; k0 += 64) {
    { // stage A 64x64 f16: two 16B chunks per thread
      const int m = tid >> 2, kq = (tid & 3) * 16;
      const f16* gp = Ctx + (size_t)(m0 + m) * DD + k0 + kq;
      f16* lp = &As[m * LDA + kq];
#if USE_ASYNC_COPY
      asm volatile("global_load_async_to_lds_b128 %0, %1, off"
                   :: "v"((unsigned)(uintptr_t)lp),
                      "v"((unsigned long long)(uintptr_t)gp)
                   : "memory");
      asm volatile("global_load_async_to_lds_b128 %0, %1, off"
                   :: "v"((unsigned)(uintptr_t)(lp + 8)),
                      "v"((unsigned long long)(uintptr_t)(gp + 8))
                   : "memory");
#else
      *(v8h*)lp       = *(const v8h*)gp;
      *(v8h*)(lp + 8) = *(const v8h*)(gp + 8);
#endif
    }
    { // stage B 64x64 column-major from f32 weights
      const int n = tid & 63, kb4 = tid >> 6;
      #pragma unroll
      for (int j = 0; j < 8; ++j) {
        const int k = 2 * (kb4 + 4 * j);
        const float lo = Wo[(size_t)(k0 + k) * DD + n0 + n];
        const float hi = Wo[(size_t)(k0 + k + 1) * DD + n0 + n];
        h2 p; p.x = (f16)lo; p.y = (f16)hi;
        *(h2*)&Bs[n * LDA + k] = p;
      }
    }
#if USE_ASYNC_COPY
    asm volatile("s_wait_asynccnt 0x0" ::: "memory");
#endif
    __syncthreads();

    #pragma unroll
    for (int ks = 0; ks < 2; ++ks) {
      V16 a;
      const f16* ap = &As[(wm * 16 + col) * LDA + ks * 32];
      a.h[0] = *(const v8h*)(ap + half * 8);
      a.h[1] = *(const v8h*)(ap + 16 + half * 8);
      #pragma unroll
      for (int jn = 0; jn < 2; ++jn) {
        const f16* bp = &Bs[(wn * 16 + jn * 32 + col) * LDA + ks * 32];
        V16 b;
        b.h[0] = *(const v8h*)(bp + half * 8);
        b.h[1] = *(const v8h*)(bp + 16 + half * 8);
        if (jn == 0) acc0 = wmma16(a.v, b.v, acc0);
        else         acc1 = wmma16(a.v, b.v, acc1);
      }
    }
    __syncthreads();
  }

  #pragma unroll
  for (int jn = 0; jn < 2; ++jn) {
    const v8f& acc = jn ? acc1 : acc0;
    #pragma unroll
    for (int r = 0; r < 8; ++r) {
      const int gm = m0 + wm * 16 + 8 * half + r;
      const int gn = n0 + wn * 16 + jn * 32 + col;
      Y[(size_t)gm * DD + gn] = acc[r] + bo[gn] + X[(size_t)gm * DD + gn];
    }
  }
}

// ---------------------------------------------------------------------------
// Kernel 4: LayerNorm over last dim (1024): exactly one float4 per thread.
// ---------------------------------------------------------------------------
__global__ __launch_bounds__(256)
void layernorm(const float* __restrict__ Xr, const float* __restrict__ gamma,
               const float* __restrict__ beta, float* __restrict__ out)
{
  __shared__ float wsum[8], wsq[8];
  const int row = blockIdx.x;
  const float* x = Xr + (size_t)row * DD;
  const int i4 = threadIdx.x * 4;

  const float4 v = *(const float4*)(x + i4);
  float s  = v.x + v.y + v.z + v.w;
  float ss = v.x * v.x + v.y * v.y + v.z * v.z + v.w * v.w;

  #pragma unroll
  for (int m = 1; m < 32; m <<= 1) {
    s  += __shfl_xor(s,  m, 32);
    ss += __shfl_xor(ss, m, 32);
  }
  const int wv = threadIdx.x >> 5, ln = threadIdx.x & 31;
  if (ln == 0) { wsum[wv] = s; wsq[wv] = ss; }
  __syncthreads();

  float ts = 0.f, tss = 0.f;
  #pragma unroll
  for (int i = 0; i < 8; ++i) { ts += wsum[i]; tss += wsq[i]; }
  const float mean = ts * (1.0f / DD);
  const float var  = tss * (1.0f / DD) - mean * mean;
  const float inv  = rsqrtf(var + 1e-5f);

  const float4 g = *(const float4*)(gamma + i4);
  const float4 be = *(const float4*)(beta + i4);
  float4 o;
  o.x = g.x * (v.x - mean) * inv + be.x;
  o.y = g.y * (v.y - mean) * inv + be.y;
  o.z = g.z * (v.z - mean) * inv + be.z;
  o.w = g.w * (v.w - mean) * inv + be.w;
  *(float4*)(out + (size_t)row * DD + i4) = o;
}

// ---------------------------------------------------------------------------
extern "C" void kernel_launch(void* const* d_in, const int* in_sizes, int n_in,
                              void* d_out, int out_size, void* d_ws, size_t ws_size,
                              hipStream_t stream)
{
  const float* X     = (const float*)d_in[0];
  const float* mask  = (const float*)d_in[1];
  const float* Wq    = (const float*)d_in[2];
  const float* bq    = (const float*)d_in[3];
  const float* Wk    = (const float*)d_in[4];
  const float* bk    = (const float*)d_in[5];
  const float* Wv    = (const float*)d_in[6];
  const float* bv    = (const float*)d_in[7];
  const float* Wo    = (const float*)d_in[8];
  const float* bo    = (const float*)d_in[9];
  const float* gamma = (const float*)d_in[10];
  const float* beta  = (const float*)d_in[11];

  float* out        = (float*)d_out;                 // [B,S,D] f32
  float* out_scores = out + (size_t)MM * DD;         // [B,H,S,S] f32

  char* ws = (char*)d_ws;
  f16*   Q   = (f16*)(ws);
  f16*   Kf  = (f16*)(ws + (((size_t)8)  << 20));
  f16*   Vt  = (f16*)(ws + (((size_t)16) << 20));
  f16*   Ctx = (f16*)(ws + (((size_t)24) << 20));
  float* Y   = (float*)(ws + (((size_t)32) << 20));

  qkv_gemm<<<dim3(MM / 64, DD / 64, 3), 256, 0, stream>>>(
      X, Wq, bq, Wk, bk, Wv, bv, Q, Kf, Vt);
  attn_fa<<<dim3(8 * HH, SS / 128), 256, 0, stream>>>(
      Q, Kf, Vt, mask, out_scores, Ctx);
  out_gemm<<<dim3(MM / 64, DD / 64), 256, 0, stream>>>(Ctx, Wo, bo, X, Y);
  layernorm<<<MM, 256, 0, stream>>>(Y, gamma, beta, out);
}
